// JSDivKNNSoftmaxLoss_35510789604101
// MI455X (gfx1250) — compile-verified, matrix operand order checked by reading, not run on hardware
//
#include <hip/hip_runtime.h>
#include <hip/hip_bf16.h>
#include <math.h>

// Problem constants (match reference)
static constexpr int    kN   = 4096;
static constexpr int    kD   = 512;
static constexpr int    kNB  = 4;
static constexpr int    kBD  = 128;
static constexpr int    kK   = 100;     // KNN_K
static constexpr float  kALPHA = 40.0f;
static constexpr float  kBETA  = 40.0f;

typedef __attribute__((ext_vector_type(16))) _Float16 v16h;
typedef __attribute__((ext_vector_type(8)))  _Float16 v8h;
typedef __attribute__((ext_vector_type(8)))  float    v8f;

// ---------------- workspace layout -------------------------------------------
// [0)                 Xh   : _Float16 [NB][N][BD]   (4 MB)
// [XH_BYTES)          Dist : float    [NB][N][N]    (256 MB)
// [ACC_OFF)           accf : float[32]  -- 0..3 loss_num, 4..7 haspos,
//                                          8..11 prec, 12..15 same_sum,
//                                          16..19 diff_sum, 20..25 kl[6]
// [ACC_OFF+128)       accu : uint[8]    -- 0 cnt_same, 1 cnt_diff
static constexpr size_t XH_BYTES   = (size_t)kNB * kN * kBD * sizeof(_Float16);
static constexpr size_t DIST_OFF   = XH_BYTES;
static constexpr size_t DIST_BYTES = (size_t)kNB * kN * kN * sizeof(float);
static constexpr size_t ACC_OFF    = DIST_OFF + DIST_BYTES;

// ---------------- accumulator init -------------------------------------------
__global__ void init_acc_kernel(unsigned int* acc_words) {
    int t = threadIdx.x;
    if (t < 40) acc_words[t] = 0u;
}

// ---------------- normalize: fp32 -> unit-norm f16 per (branch,row) ----------
__global__ void __launch_bounds__(kBD)
normalize_kernel(const float* __restrict__ x, const int* __restrict__ bidx,
                 _Float16* __restrict__ xh) {
    const int i = blockIdx.x;      // row
    const int b = blockIdx.y;      // branch
    const int t = threadIdx.x;     // dim within branch (0..127)
    const int col = bidx[b * kBD + t];
    const float v = x[(size_t)i * kD + col];

    __shared__ float red[kBD];
    red[t] = v * v;
    __syncthreads();
    for (int s = kBD / 2; s > 0; s >>= 1) {
        if (t < s) red[t] += red[t + s];
        __syncthreads();
    }
    const float inv = rsqrtf(red[0]);
    xh[((size_t)b * kN + i) * kBD + t] = (_Float16)(v * inv);
}

// ---------------- WMMA Gram + distance ---------------------------------------
// A (16x32 f16, MxK) per-lane layout (ISA 7.12.2): lane<16 -> K 0..7 & 16..23,
// lane>=16 -> K 8..15 & 24..31, row M = lane&15.  B = X^T (KxN), so B tiles of
// the Gram load identically from rows of X.
__device__ inline v16h load_ab_tile(const _Float16* __restrict__ rowp,
                                    int k, int ksel) {
    union { v16h v; v8h h[2]; } u;
    u.h[0] = *(const v8h*)(rowp + k + ksel);        // K = k+ksel   .. +7
    u.h[1] = *(const v8h*)(rowp + k + ksel + 16);   // K = k+ksel+16.. +23
    return u.v;
}

__device__ inline void store_dist_tile(float* __restrict__ dist, size_t base,
                                       int i0, int j0, int mOff, int nOff,
                                       const v8f& a) {
    #pragma unroll
    for (int v = 0; v < 8; ++v) {
        const float g  = a[v];
        const float d2 = fmaxf(2.0f - 2.0f * g, 1e-12f);
        dist[base + (size_t)(i0 + mOff + v) * kN + (j0 + nOff)] = sqrtf(d2);
    }
}

__global__ void __launch_bounds__(128)
gram_dist_kernel(const _Float16* __restrict__ xh, float* __restrict__ dist) {
    const int b    = blockIdx.z;
    const int lane = threadIdx.x & 31;
    const int wave = threadIdx.x >> 5;
    const int i0 = blockIdx.y * 64 + (wave >> 1) * 32;
    const int j0 = blockIdx.x * 64 + (wave & 1) * 32;

    const _Float16* Xb = xh + (size_t)b * kN * kBD;
    const int mrow = lane & 15;
    const int ksel = (lane >> 4) * 8;   // 0 for lanes 0..15, 8 for 16..31

    const _Float16* ar0 = Xb + (size_t)(i0 + mrow)      * kBD;
    const _Float16* ar1 = Xb + (size_t)(i0 + 16 + mrow) * kBD;
    const _Float16* br0 = Xb + (size_t)(j0 + mrow)      * kBD;
    const _Float16* br1 = Xb + (size_t)(j0 + 16 + mrow) * kBD;

    v8f acc00 = {}, acc01 = {}, acc10 = {}, acc11 = {};

    #pragma unroll
    for (int k = 0; k < kBD; k += 32) {
        const v16h a0 = load_ab_tile(ar0, k, ksel);
        const v16h a1 = load_ab_tile(ar1, k, ksel);
        const v16h b0 = load_ab_tile(br0, k, ksel);
        const v16h b1 = load_ab_tile(br1, k, ksel);
        acc00 = __builtin_amdgcn_wmma_f32_16x16x32_f16(false, a0, false, b0,
                                                       (short)0, acc00, false, false);
        acc01 = __builtin_amdgcn_wmma_f32_16x16x32_f16(false, a0, false, b1,
                                                       (short)0, acc01, false, false);
        acc10 = __builtin_amdgcn_wmma_f32_16x16x32_f16(false, a1, false, b0,
                                                       (short)0, acc10, false, false);
        acc11 = __builtin_amdgcn_wmma_f32_16x16x32_f16(false, a1, false, b1,
                                                       (short)0, acc11, false, false);
    }

    const size_t base = (size_t)b * kN * kN;
    const int mOff = (lane >> 4) * 8;   // C/D layout: M = vgpr + 8*(lane/16)
    const int nOff = lane & 15;         //             N = lane%16
    store_dist_tile(dist, base, i0,      j0,      mOff, nOff, acc00);
    store_dist_tile(dist, base, i0,      j0 + 16, mOff, nOff, acc01);
    store_dist_tile(dist, base, i0 + 16, j0,      mOff, nOff, acc10);
    store_dist_tile(dist, base, i0 + 16, j0 + 16, mOff, nOff, acc11);
}

// ---------------- top-100 KNN softmax loss per (branch,row) ------------------
__global__ void __launch_bounds__(256)
knn_kernel(const float* __restrict__ dist, const int* __restrict__ targets,
           float* __restrict__ accf) {
    const int r = blockIdx.x;
    const int b = blockIdx.y;
    const int t = threadIdx.x;
    const float* row = dist + ((size_t)b * kN + r) * kN;

    __shared__ float rowv[kN];           // 16 KB
    __shared__ float rmin[256];
    __shared__ int   rloc[256];
    __shared__ float kd[kK];
    __shared__ int   ksame[kK];

    const int tcls = targets[r];
    for (int c = t; c < kN; c += 256) rowv[c] = (c == r) ? 1e9f : row[c];
    __syncthreads();

    for (int sel = 0; sel < kK; ++sel) {
        float mv = 1e30f; int mi = t;
        for (int c = t; c < kN; c += 256) {
            const float v = rowv[c];
            if (v < mv) { mv = v; mi = c; }
        }
        rmin[t] = mv; rloc[t] = mi;
        __syncthreads();
        for (int s = 128; s > 0; s >>= 1) {
            if (t < s) {
                const float ov = rmin[t + s]; const int oi = rloc[t + s];
                if (ov < rmin[t] || (ov == rmin[t] && oi < rloc[t])) {
                    rmin[t] = ov; rloc[t] = oi;
                }
            }
            __syncthreads();
        }
        if (t == 0) {
            const int idx = rloc[0];
            kd[sel]    = rmin[0];
            ksame[sel] = (targets[idx] == tcls) ? 1 : 0;
            rowv[idx]  = 1e9f;   // consume
        }
        __syncthreads();
    }

    if (t == 0) {
        const float mx = -kALPHA * kd[0];   // kd ascending -> max logit
        float Z = 0.0f, pos = 0.0f; int hp = 0;
        for (int q = 0; q < kK; ++q) {
            const float e = expf(-kALPHA * kd[q] - mx);
            Z += e;
            if (ksame[q]) { pos += e; hp = 1; }
        }
        const float pos_p = pos / Z;
        const float per   = -logf(fmaxf(pos_p, 1e-8f));
        if (hp) {
            atomicAdd(&accf[0 + b], per);
            atomicAdd(&accf[4 + b], 1.0f);
        }
        atomicAdd(&accf[8 + b], (float)ksame[0]);   // 1-NN precision
    }
}

// ---------------- same/diff class distance sums ------------------------------
__global__ void __launch_bounds__(256)
posneg_kernel(const float* __restrict__ dist, const int* __restrict__ targets,
              float* __restrict__ accf, unsigned int* __restrict__ accu) {
    const int r = blockIdx.x;
    const int b = blockIdx.y;
    const int t = threadIdx.x;
    const float* row = dist + ((size_t)b * kN + r) * kN;
    const int tc = targets[r];

    float ss = 0.0f, sd = 0.0f;
    unsigned int cs = 0u, cd = 0u;
    for (int c = t; c < kN; c += 256) {
        if (c == r) continue;
        const float d = row[c];
        if (targets[c] == tc) { ss += d; cs++; } else { sd += d; cd++; }
    }

    __shared__ float s1[256], s2[256];
    __shared__ unsigned int u1[256], u2[256];
    s1[t] = ss; s2[t] = sd; u1[t] = cs; u2[t] = cd;
    __syncthreads();
    for (int s = 128; s > 0; s >>= 1) {
        if (t < s) { s1[t] += s1[t+s]; s2[t] += s2[t+s]; u1[t] += u1[t+s]; u2[t] += u2[t+s]; }
        __syncthreads();
    }
    if (t == 0) {
        atomicAdd(&accf[12 + b], s1[0]);
        atomicAdd(&accf[16 + b], s2[0]);
        if (b == 0) { atomicAdd(&accu[0], u1[0]); atomicAdd(&accu[1], u2[0]); }
    }
}

// ---------------- JS divergence over 6 branch pairs, per row -----------------
__global__ void __launch_bounds__(256)
js_kernel(const float* __restrict__ dist, float* __restrict__ accf) {
    const int r = blockIdx.x;
    const int t = threadIdx.x;

    __shared__ float red[256];
    __shared__ float mx[kNB], lz[kNB];

    // Per-branch stable log-sum-exp of logits = -BETA * dist over off-diag cols
    for (int b = 0; b < kNB; ++b) {
        const float* row = dist + ((size_t)b * kN + r) * kN;
        float m = -1e30f;
        for (int c = t; c < kN; c += 256) {
            if (c == r) continue;
            m = fmaxf(m, -kBETA * row[c]);
        }
        red[t] = m; __syncthreads();
        for (int s = 128; s > 0; s >>= 1) {
            if (t < s) red[t] = fmaxf(red[t], red[t+s]);
            __syncthreads();
        }
        const float mb = red[0];
        __syncthreads();
        float z = 0.0f;
        for (int c = t; c < kN; c += 256) {
            if (c == r) continue;
            z += expf(-kBETA * row[c] - mb);
        }
        red[t] = z; __syncthreads();
        for (int s = 128; s > 0; s >>= 1) {
            if (t < s) red[t] += red[t+s];
            __syncthreads();
        }
        if (t == 0) { mx[b] = mb; lz[b] = logf(red[0]); }
        __syncthreads();
    }

    int pi = 0;
    for (int bi = 0; bi < kNB - 1; ++bi) {
        for (int bj = bi + 1; bj < kNB; ++bj, ++pi) {
            const float* ri = dist + ((size_t)bi * kN + r) * kN;
            const float* rj = dist + ((size_t)bj * kN + r) * kN;
            float acc = 0.0f;
            for (int c = t; c < kN; c += 256) {
                if (c == r) continue;
                const float li = -kBETA * ri[c] - mx[bi] - lz[bi];
                const float lj = -kBETA * rj[c] - mx[bj] - lz[bj];
                const float si = expf(li), sj = expf(lj);
                const float m  = 0.5f * (si + sj);
                if (m > 0.0f) {
                    const float lm = logf(m);
                    acc += m * (lm - li) + m * (lm - lj);
                }
            }
            red[t] = acc; __syncthreads();
            for (int s = 128; s > 0; s >>= 1) {
                if (t < s) red[t] += red[t+s];
                __syncthreads();
            }
            if (t == 0) atomicAdd(&accf[20 + pi], red[0]);
            __syncthreads();
        }
    }
}

// ---------------- final scalar combine ---------------------------------------
__global__ void finalize_kernel(const float* __restrict__ accf,
                                const unsigned int* __restrict__ accu,
                                float* __restrict__ out) {
    float knn = 0.0f, acc = 0.0f, pd = 0.0f, nd = 0.0f;
    const float cs = (float)accu[0];
    const float cd = (float)accu[1];
    for (int b = 0; b < kNB; ++b) {
        knn += accf[0 + b] / fmaxf(accf[4 + b], 1.0f);
        acc += accf[8 + b] / (float)kN;
        pd  += accf[12 + b] / cs;
        nd  += accf[16 + b] / cd;
    }
    knn /= (float)kNB; acc /= (float)kNB; pd /= (float)kNB; nd /= (float)kNB;
    float js = 0.0f;
    for (int p = 0; p < 6; ++p) js += (0.5f / (float)kN) * accf[20 + p];
    js /= 6.0f;
    out[0] = knn + expf(-js);   // GAMA=1, SIGMA=1
    out[1] = js;
    out[2] = acc;
    out[3] = pd;
    out[4] = nd;
}

// ---------------- launcher ---------------------------------------------------
extern "C" void kernel_launch(void* const* d_in, const int* in_sizes, int n_in,
                              void* d_out, int out_size, void* d_ws, size_t ws_size,
                              hipStream_t stream) {
    (void)in_sizes; (void)n_in; (void)out_size; (void)ws_size;

    const float* x       = (const float*)d_in[0];   // (N, D)
    const int*   targets = (const int*)d_in[1];     // (N,)
    const int*   bidx    = (const int*)d_in[2];     // (NB, BD)
    float*       out     = (float*)d_out;           // 5 scalars

    char* ws = (char*)d_ws;
    _Float16*     xh   = (_Float16*)ws;
    float*        dist = (float*)(ws + DIST_OFF);
    float*        accf = (float*)(ws + ACC_OFF);
    unsigned int* accu = (unsigned int*)(ws + ACC_OFF + 32 * sizeof(float));

    init_acc_kernel<<<1, 64, 0, stream>>>((unsigned int*)(ws + ACC_OFF));

    normalize_kernel<<<dim3(kN, kNB), kBD, 0, stream>>>(x, bidx, xh);

    gram_dist_kernel<<<dim3(kN / 64, kN / 64, kNB), 128, 0, stream>>>(xh, dist);

    knn_kernel<<<dim3(kN, kNB), 256, 0, stream>>>(dist, targets, accf);

    posneg_kernel<<<dim3(kN, kNB), 256, 0, stream>>>(dist, targets, accf, accu);

    js_kernel<<<dim3(kN), 256, 0, stream>>>(dist, accf);

    finalize_kernel<<<1, 1, 0, stream>>>(accf, accu, out);
}
